// InnerSelfAttention_78417512891096
// MI455X (gfx1250) — matrix operand order repaired and checked by reading, not compiled, not run on hardware
//
#include <hip/hip_runtime.h>

// ---------------------------------------------------------------------------
// CDNA5 (gfx1250) sliding-window causal attention, wave32 + WMMA bf16 + TDM.
//   B=2, S=2048, D=1024, H=16, HD=64, WINDOW=256
// Pipeline:
//   1) qkv_proj_kernel : Q,K = X@W^T -> bf16 [B,H,S,HD]; V -> bf16 [B,H,HD,S]
//      (32x64 output tile per wave: 8 WMMA per k-step, 3 b128 loads/WMMA)
//   2) flash_attn_kernel: 16-query tile/wave, 32-key chunks; K (1-D tile) and
//      V (2-D strided tile) double-buffered into LDS via tensor_load_to_lds,
//      online softmax in fp32, WMMA PV
//   3) out_proj_kernel : ctx(bf16) @ Wo^T + bo -> fp32 out (32x64 tile/wave)
// Workspace: 4 x 8MB bf16 buffers (Q, K, Vt, ctx) = 32 MB.
// ---------------------------------------------------------------------------

typedef __bf16        v16bf __attribute__((ext_vector_type(16)));
typedef float         v8f   __attribute__((ext_vector_type(8)));
typedef unsigned int  u32x4 __attribute__((ext_vector_type(4)));
typedef unsigned int  u32x8 __attribute__((ext_vector_type(8)));
typedef float         f32x4 __attribute__((ext_vector_type(4)));

#define NB    2
#define NS    2048
#define ND    1024
#define NH    16
#define NHD   64
#define NEGF  (-3.402823466e38f)

union Frag16 {               // 16 bf16 values = 8 VGPRs (wave32 WMMA operand)
    v16bf  v;
    __bf16 h[16];
    u32x4  q[2];
};

__device__ __forceinline__ v8f wmma_bf16(v16bf a, v16bf b, v8f c) {
    // D(16x16 f32) = A(16x32 bf16) * B(32x16 bf16) + C
    return __builtin_amdgcn_wmma_f32_16x16x32_bf16(
        /*neg_a=*/false, a, /*neg_b=*/false, b,
        /*c_mod=*/(short)0, c, /*reuse_a=*/false, /*reuse_b=*/false);
}

__device__ __forceinline__ v8f v8f_zero() {
    v8f z;
#pragma unroll
    for (int i = 0; i < 8; ++i) z[i] = 0.0f;
    return z;
}

// TDM: load a (tile0 x tile1) tile of bf16 from global `gaddr` (row stride
// `stride0` elements) into LDS at byte-offset `lds_byte`.  D# built per the
// CDNA5 ISA Tensor DMA Descriptor layout (group0: 4 SGPRs, group1: 8 SGPRs).
__device__ __forceinline__ void tdm_load(unsigned long long gaddr,
                                         unsigned lds_byte,
                                         unsigned tile0, unsigned tile1,
                                         unsigned tdim0, unsigned tdim1,
                                         unsigned stride0) {
    u32x4 g0;
    g0[0] = 1u;                                   // count=1, user descriptor
    g0[1] = lds_byte;                             // lds_addr (bytes)
    g0[2] = (unsigned)gaddr;                      // global_addr[31:0]
    g0[3] = (unsigned)((gaddr >> 32) & 0x01FFFFFFu) | (2u << 30); // type=2
    u32x8 g1;
    g1[0] = (1u << 16);                           // data_size=1 -> 2 bytes
    g1[1] = (tdim0 & 0xFFFFu) << 16;              // tensor_dim0[15:0]
    g1[2] = (tdim0 >> 16) | ((tdim1 & 0xFFFFu) << 16);
    g1[3] = (tdim1 >> 16) | (tile0 << 16);        // tile_dim0
    g1[4] = tile1;                                // tile_dim1 (tile_dim2 = 0)
    g1[5] = stride0;                              // tensor_dim0_stride lo
    g1[6] = 0u;
    g1[7] = 0u;
    asm volatile("tensor_load_to_lds %0, %1" :: "s"(g0), "s"(g1) : "memory");
}

// ---------------------------------------------------------------------------
// Kernel 1: fused Q/K/V projection.  One wave = 32x64 output tile
// (2 M-tiles x 4 N-tiles, 8 accumulators).  K-loop: 1024 in steps of 32.
// Tiles: sel(3) x mt(128) x ng(16)  ->  6144 waves / 8 per block.
// ---------------------------------------------------------------------------
__global__ __launch_bounds__(256)
void qkv_proj_kernel(const float* __restrict__ x,
                     const float* __restrict__ wqp,
                     const float* __restrict__ wkp,
                     const float* __restrict__ wvp,
                     __bf16* __restrict__ qo,   // [B,H,S,HD]
                     __bf16* __restrict__ ko,   // [B,H,S,HD]
                     __bf16* __restrict__ vto)  // [B,H,HD,S]
{
    const int wid  = blockIdx.x * 8 + (threadIdx.x >> 5);
    const int lane = threadIdx.x & 31;
    const int lr   = lane & 15;
    const int hi   = lane >> 4;
    const int ng   = wid & 15;
    const int mt   = (wid >> 4) & 127;
    const int sel  = wid >> 11;                  // 0=Q, 1=K, 2=V
    const float* w = (sel == 0) ? wqp : ((sel == 1) ? wkp : wvp);
    const int m0 = mt * 32, n0 = ng * 64;

    v8f acc[2][4];
#pragma unroll
    for (int m = 0; m < 2; ++m)
#pragma unroll
        for (int n = 0; n < 4; ++n) acc[m][n] = v8f_zero();

    for (int k0 = 0; k0 < ND; k0 += 32) {
        Frag16 af[2];
#pragma unroll
        for (int m = 0; m < 2; ++m) {
            const f32x4* ap =
                (const f32x4*)(x + (m0 + m * 16 + lr) * ND + k0 + hi * 16);
            f32x4 a0 = ap[0], a1 = ap[1], a2 = ap[2], a3 = ap[3];
#pragma unroll
            for (int j = 0; j < 4; ++j) {
                af[m].h[j]      = (__bf16)a0[j];
                af[m].h[4 + j]  = (__bf16)a1[j];
                af[m].h[8 + j]  = (__bf16)a2[j];
                af[m].h[12 + j] = (__bf16)a3[j];
            }
        }
#pragma unroll
        for (int n = 0; n < 4; ++n) {
            Frag16 bf;
            const f32x4* bp =
                (const f32x4*)(w + (n0 + n * 16 + lr) * ND + k0 + hi * 16);
            f32x4 b0 = bp[0], b1 = bp[1], b2 = bp[2], b3 = bp[3];
#pragma unroll
            for (int j = 0; j < 4; ++j) {
                bf.h[j]      = (__bf16)b0[j];
                bf.h[4 + j]  = (__bf16)b1[j];
                bf.h[8 + j]  = (__bf16)b2[j];
                bf.h[12 + j] = (__bf16)b3[j];
            }
            acc[0][n] = wmma_bf16(af[0].v, bf.v, acc[0][n]);
            acc[1][n] = wmma_bf16(af[1].v, bf.v, acc[1][n]);
        }
    }

    // Branch-free scatter into head-split layouts.
    __bf16* dst = (sel == 0) ? qo : ((sel == 1) ? ko : vto);
    const bool isV = (sel == 2);
    const int hI = n0 >> 6;                      // all 4 n-tiles in one head
#pragma unroll
    for (int m = 0; m < 2; ++m)
#pragma unroll
        for (int n = 0; n < 4; ++n) {
            const int dI = n * 16 + lr;
#pragma unroll
            for (int r = 0; r < 8; ++r) {
                const int row = m0 + m * 16 + r + hi * 8;   // flattened (b,s)
                const int bb  = row >> 11, ss = row & 2047;
                const int bh  = bb * NH + hI;
                const int idx = isV ? ((bh * NHD + dI) * NS + ss)
                                    : ((bh * NS + ss) * NHD + dI);
                dst[idx] = (__bf16)acc[m][n][r];
            }
        }
}

// ---------------------------------------------------------------------------
// Kernel 2: sliding-window flash attention.  2 waves/block; one wave = 16
// queries of one (b,h).  K chunks (32x64 bf16, contiguous) and V chunks
// (64 rows x 32 keys, stride 2048) are double-buffered into LDS by the
// Tensor Data Mover; per-wave TDM ops retire in order, so after issuing the
// next chunk's pair, s_wait_tensorcnt<=2 guarantees the current pair landed.
// ---------------------------------------------------------------------------
__global__ __launch_bounds__(64)
void flash_attn_kernel(const __bf16* __restrict__ q,   // [B,H,S,HD]
                       const __bf16* __restrict__ k,   // [B,H,S,HD]
                       const __bf16* __restrict__ vt,  // [B,H,HD,S]
                       __bf16* __restrict__ ctx)       // [B,S,D]
{
    __shared__ __attribute__((aligned(16))) __bf16 pbuf[2 * 16 * 32];
    __shared__ __attribute__((aligned(16))) __bf16 kstage[2][2][32 * NHD];
    __shared__ __attribute__((aligned(16))) __bf16 vstage[2][2][NHD * 32];
    const int lwid = threadIdx.x >> 5;
    const int gw   = blockIdx.x * 2 + lwid;
    const int lane = threadIdx.x & 31;
    const int lr   = lane & 15;
    const int hi   = lane >> 4;
    const int b  = gw >> 11;
    const int h  = (gw >> 7) & 15;
    const int qt = gw & 127;
    const int q0 = qt * 16;
    const int bh = b * NH + h;

    __bf16* pb = pbuf + lwid * 512;
    const __bf16* qb = q  + bh * NS * NHD;
    const __bf16* kb = k  + bh * NS * NHD;
    const __bf16* vb = vt + bh * NHD * NS;
    const unsigned ldsK[2] = {
        (unsigned)(unsigned long long)&kstage[lwid][0][0],
        (unsigned)(unsigned long long)&kstage[lwid][1][0]
    };
    const unsigned ldsV[2] = {
        (unsigned)(unsigned long long)&vstage[lwid][0][0],
        (unsigned)(unsigned long long)&vstage[lwid][1][0]
    };

    // Q A-fragments for d=[0,32) and d=[32,64)
    Frag16 aQ0, aQ1;
    {
        const u32x4* qp = (const u32x4*)(qb + (q0 + lr) * NHD + hi * 16);
        aQ0.q[0] = qp[0]; aQ0.q[1] = qp[1];
        aQ1.q[0] = qp[4]; aQ1.q[1] = qp[5];
    }

    v8f acc[4];
#pragma unroll
    for (int n = 0; n < 4; ++n) acc[n] = v8f_zero();
    float mR[8], lR[8];
#pragma unroll
    for (int r = 0; r < 8; ++r) { mR[r] = NEGF; lR[r] = 0.0f; }

    const int cbeg = (q0 >= 256) ? ((q0 - 255) & ~31) : 0;
    const int cend = (q0 + 15) & ~31;

    // Prime the TDM pipeline with chunk cbeg (K 1-D tile, V 2-D tile).
    tdm_load((unsigned long long)(const void*)(kb + (size_t)cbeg * NHD),
             ldsK[0], 32 * NHD, 1, (unsigned)(NS - cbeg) * NHD, 1, 32 * NHD);
    tdm_load((unsigned long long)(const void*)(vb + cbeg),
             ldsV[0], 32, NHD, (unsigned)(NS - cbeg), NHD, NS);

    int buf = 0;
    for (int c0 = cbeg; c0 <= cend; c0 += 32) {
        // Issue TDM pair for the next chunk (clamped re-issue of the last
        // chunk is harmless), then wait for the current chunk's pair.
        const int c0n = (c0 + 32 <= cend) ? (c0 + 32) : cend;
        tdm_load((unsigned long long)(const void*)(kb + (size_t)c0n * NHD),
                 ldsK[buf ^ 1], 32 * NHD, 1,
                 (unsigned)(NS - c0n) * NHD, 1, 32 * NHD);
        tdm_load((unsigned long long)(const void*)(vb + c0n),
                 ldsV[buf ^ 1], 32, NHD, (unsigned)(NS - c0n), NHD, NS);
        __builtin_amdgcn_s_wait_tensorcnt((short)2);
        asm volatile("" ::: "memory");
        const __bf16* kc = &kstage[lwid][buf][0];
        const __bf16* vc = &vstage[lwid][buf][0];
        buf ^= 1;

        // ---- scores for the two 16-key halves of this chunk (K from LDS) ----
        v8f s0, s1;
        {
            Frag16 b0, b1;
            const u32x4* kp = (const u32x4*)(kc + lr * NHD + hi * 16);
            b0.q[0] = kp[0]; b0.q[1] = kp[1];      // d = hi*16 .. +15
            b1.q[0] = kp[4]; b1.q[1] = kp[5];      // d = 32+hi*16 .. +15
            v8f z = v8f_zero();
            s0 = wmma_bf16(aQ0.v, b0.v, z);
            s0 = wmma_bf16(aQ1.v, b1.v, s0);
        }
        {
            Frag16 b0, b1;
            const u32x4* kp = (const u32x4*)(kc + (16 + lr) * NHD + hi * 16);
            b0.q[0] = kp[0]; b0.q[1] = kp[1];
            b1.q[0] = kp[4]; b1.q[1] = kp[5];
            v8f z = v8f_zero();
            s1 = wmma_bf16(aQ0.v, b0.v, z);
            s1 = wmma_bf16(aQ1.v, b1.v, s1);
        }

        // ---- mask + online softmax update (per owned row) ----
#pragma unroll
        for (int r = 0; r < 8; ++r) {
            const int i  = q0 + r + hi * 8;
            const int j0 = c0 + lr;
            const int j1 = c0 + 16 + lr;
            const bool v0 = (j0 <= i) && (j0 >= i - 255);
            const bool v1 = (j1 <= i) && (j1 >= i - 255);
            float x0 = v0 ? s0[r] : NEGF;
            float x1 = v1 ? s1[r] : NEGF;
            float mx = fmaxf(x0, x1);
            mx = fmaxf(mx, __shfl_xor(mx, 1, 32));
            mx = fmaxf(mx, __shfl_xor(mx, 2, 32));
            mx = fmaxf(mx, __shfl_xor(mx, 4, 32));
            mx = fmaxf(mx, __shfl_xor(mx, 8, 32));
            const float mn    = fmaxf(mR[r], mx);
            const float alpha = __expf(mR[r] - mn);
            const float p0 = v0 ? __expf(x0 - mn) : 0.0f;
            const float p1 = v1 ? __expf(x1 - mn) : 0.0f;
            float rs = p0 + p1;
            rs += __shfl_xor(rs, 1, 32);
            rs += __shfl_xor(rs, 2, 32);
            rs += __shfl_xor(rs, 4, 32);
            rs += __shfl_xor(rs, 8, 32);
            lR[r] = lR[r] * alpha + rs;
            mR[r] = mn;
#pragma unroll
            for (int n = 0; n < 4; ++n) acc[n][r] *= alpha;
            // stage P (bf16) to LDS in row-major [16][32] for layout transpose
            const int prow = (r + hi * 8) * 32;
            pb[prow + lr]      = (__bf16)p0;
            pb[prow + 16 + lr] = (__bf16)p1;
        }

        // ---- P as A-fragment (LDS read in A layout) ----
        Frag16 aP;
        {
            const u32x4* pp = (const u32x4*)(pb + lr * 32 + hi * 16);
            aP.q[0] = pp[0];      // k = hi*16 .. +7
            aP.q[1] = pp[1];      // k = hi*16+8 .. +15
        }

        // ---- PV: 4 output d-tiles, V tile from LDS ([64 dims][32 keys]) ----
#pragma unroll
        for (int n = 0; n < 4; ++n) {
            Frag16 bv;
            const u32x4* vp =
                (const u32x4*)(vc + (n * 16 + lr) * 32 + hi * 16);
            bv.q[0] = vp[0]; bv.q[1] = vp[1];
            acc[n] = wmma_bf16(aP.v, bv.v, acc[n]);
        }
    }

    // ---- normalize and write ctx [B,S,D] (bf16) ----
#pragma unroll
    for (int r = 0; r < 8; ++r) {
        const float inv = 1.0f / lR[r];
        const int row = q0 + r + hi * 8;
        __bf16* cp = ctx + (b * NS + row) * ND + h * NHD;
#pragma unroll
        for (int n = 0; n < 4; ++n) cp[n * 16 + lr] = (__bf16)(acc[n][r] * inv);
    }
}

// ---------------------------------------------------------------------------
// Kernel 3: output projection  out = ctx @ Wo^T + bo   (fp32 out)
// One wave = 32x64 output tile (8 accumulators).
// ---------------------------------------------------------------------------
__global__ __launch_bounds__(256)
void out_proj_kernel(const __bf16* __restrict__ ctx,  // bf16 [B,S,D]
                     const float* __restrict__ wo,
                     const float* __restrict__ bo,
                     float* __restrict__ out)
{
    const int wid  = blockIdx.x * 8 + (threadIdx.x >> 5);
    const int lane = threadIdx.x & 31;
    const int lr   = lane & 15;
    const int hi   = lane >> 4;
    const int ng   = wid & 15;
    const int mt   = wid >> 4;
    const int m0 = mt * 32, n0 = ng * 64;

    v8f acc[2][4];
#pragma unroll
    for (int m = 0; m < 2; ++m)
#pragma unroll
        for (int n = 0; n < 4; ++n) acc[m][n] = v8f_zero();

    for (int k0 = 0; k0 < ND; k0 += 32) {
        Frag16 af[2];
#pragma unroll
        for (int m = 0; m < 2; ++m) {
            const u32x4* ap =
                (const u32x4*)(ctx + (m0 + m * 16 + lr) * ND + k0 + hi * 16);
            af[m].q[0] = ap[0]; af[m].q[1] = ap[1];
        }
#pragma unroll
        for (int n = 0; n < 4; ++n) {
            Frag16 bf;
            const f32x4* bp =
                (const f32x4*)(wo + (n0 + n * 16 + lr) * ND + k0 + hi * 16);
            f32x4 b0 = bp[0], b1 = bp[1], b2 = bp[2], b3 = bp[3];
#pragma unroll
            for (int j = 0; j < 4; ++j) {
                bf.h[j]      = (__bf16)b0[j];
                bf.h[4 + j]  = (__bf16)b1[j];
                bf.h[8 + j]  = (__bf16)b2[j];
                bf.h[12 + j] = (__bf16)b3[j];
            }
            acc[0][n] = wmma_bf16(af[0].v, bf.v, acc[0][n]);
            acc[1][n] = wmma_bf16(af[1].v, bf.v, acc[1][n]);
        }
    }

#pragma unroll
    for (int n = 0; n < 4; ++n) {
        const int col  = n0 + n * 16 + lr;
        const float bias = bo[col];
#pragma unroll
        for (int m = 0; m < 2; ++m)
#pragma unroll
            for (int r = 0; r < 8; ++r)
                out[(m0 + m * 16 + r + hi * 8) * ND + col] =
                    acc[m][n][r] + bias;
    }
}

// ---------------------------------------------------------------------------
extern "C" void kernel_launch(void* const* d_in, const int* in_sizes, int n_in,
                              void* d_out, int out_size, void* d_ws,
                              size_t ws_size, hipStream_t stream)
{
    (void)in_sizes; (void)n_in; (void)out_size; (void)ws_size;
    const float* x  = (const float*)d_in[0];
    const float* wq = (const float*)d_in[1];
    const float* wk = (const float*)d_in[2];
    const float* wv = (const float*)d_in[3];
    const float* wo = (const float*)d_in[4];
    const float* bo = (const float*)d_in[5];
    float* out = (float*)d_out;

    // workspace: Q, K, Vt, ctx — each B*H*S*HD = 4,194,304 bf16 (8 MB)
    const size_t QKVN = (size_t)NB * NH * NS * NHD;
    __bf16* qws   = (__bf16*)d_ws;
    __bf16* kws   = qws + QKVN;
    __bf16* vtws  = kws + QKVN;
    __bf16* ctxws = vtws + QKVN;

    // 3*128*16 = 6144 waves / 8 per block = 768 blocks
    qkv_proj_kernel<<<dim3(768), dim3(256), 0, stream>>>(
        x, wq, wk, wv, qws, kws, vtws);
    // 2*16*128 = 4096 query tiles / 2 waves per block = 2048 blocks
    flash_attn_kernel<<<dim3(2048), dim3(64), 0, stream>>>(
        qws, kws, vtws, ctxws);
    // 128*16 = 2048 waves / 8 per block = 256 blocks
    out_proj_kernel<<<dim3(256), dim3(256), 0, stream>>>(
        ctxws, wo, bo, out);
}
